// TriplaneAutoDecoder_62637803045604
// MI455X (gfx1250) — matrix operand order repaired and checked by reading, not compiled, not run on hardware
//
#include <hip/hip_runtime.h>

typedef __attribute__((ext_vector_type(16))) _Float16 v16h;
typedef __attribute__((ext_vector_type(8)))  float    v8f;
typedef __attribute__((ext_vector_type(2)))  _Float16 h2;

#define TILE   128     // points per block
#define CDIM   32      // triplane channels
#define HDIM   128     // hidden width
#define RPLANE 256     // plane resolution
#define FPAD   40      // padded LDS row stride for 32-wide rows (80B -> bank spread)
#define HPAD   136     // padded LDS row stride for 128-wide rows (272B -> bank spread)

// ---- WMMA fragment loaders (CDNA5 16-bit layouts, ISA 7.12.2) ----
// A (16x32 f16): lanes 0-15 -> M=lane, K-halves {0..7,16..23}; lanes 16-31 -> K {8..15,24..31}
__device__ __forceinline__ v16h load_a_frag(const _Float16* __restrict__ src,
                                            int rowBase, int rowStride, int kOfs,
                                            int m, bool lo) {
  v16h a;
  const _Float16* rowp = src + (rowBase + m) * rowStride + kOfs + (lo ? 0 : 8);
#pragma unroll
  for (int i = 0; i < 8; ++i) {
    int kb = 2 * i + (i < 4 ? 0 : 8);          // 0,2,4,6 then 16,18,20,22
    h2 p = *(const h2*)(rowp + kb);
    a[2 * i]     = p[0];
    a[2 * i + 1] = p[1];
  }
  return a;
}
// B (32x16 f16): W stored [out][K]; B[k][n] = W[colBase+n][kOfs+k].
// lanes 0-15 hold K {0..7,16..23}; lanes 16-31 hold K {8..15,24..31} (N = lane%16)
__device__ __forceinline__ v16h load_b_frag(const _Float16* __restrict__ w,
                                            int colBase, int kStride, int kOfs,
                                            int n, bool lo) {
  v16h bm;
  const _Float16* rowp = w + (colBase + n) * kStride + kOfs + (lo ? 0 : 8);
#pragma unroll
  for (int i = 0; i < 8; ++i) {
    h2 p = *(const h2*)(rowp + 2 * i + (i < 4 ? 0 : 8));
    bm[2 * i]     = p[0];
    bm[2 * i + 1] = p[1];
  }
  return bm;
}

__launch_bounds__(TILE, 3)
__global__ void triplane_mlp_kernel(
    const float* __restrict__ tp,       // [B,3,32,256,256]
    const float* __restrict__ coords,   // [B,N,3]
    const float* __restrict__ w0, const float* __restrict__ b0,   // [128,32],[128]
    const float* __restrict__ w1, const float* __restrict__ b1,   // [128,128],[128]
    const float* __restrict__ w2, const float* __restrict__ b2,   // [1,128],[1]
    float* __restrict__ out,            // [B*N]
    int N, long totalPts)
{
  __shared__ _Float16 sW0[HDIM * FPAD];
  __shared__ _Float16 sW1[HDIM * HPAD];
  __shared__ float    sB0[HDIM], sB1[HDIM], sW2[HDIM];
  __shared__ _Float16 sF [TILE * FPAD];
  __shared__ _Float16 sH0[TILE * HPAD];
  __shared__ float    sOut[TILE];

  const int  t    = threadIdx.x;
  const int  lane = t & 31;
  // force wave index into an SGPR: keeps all WMMA-phase control flow scalar
  const int  wave = __builtin_amdgcn_readfirstlane(t >> 5);
  const bool lo   = (lane < 16);
  const int  m    = lane & 15;

  // ---- stage MLP weights into LDS (f32 -> f16 for WMMA, f32 for bias/w2) ----
  for (int i = t; i < HDIM * CDIM; i += TILE)
    sW0[(i / CDIM) * FPAD + (i % CDIM)] = (_Float16)w0[i];
  for (int i = t; i < HDIM * HDIM; i += TILE)
    sW1[(i >> 7) * HPAD + (i & 127)] = (_Float16)w1[i];
  if (t < HDIM) { sB0[t] = b0[t]; sB1[t] = b1[t]; sW2[t] = w2[t]; }

  // ---- phase 1: tri-plane bilinear gather (one point per thread) ----
  long gp = (long)blockIdx.x * TILE + t;
  const bool live = (gp < totalPts);
  const long gpc  = live ? gp : (totalPts - 1);
  const int  bb   = (int)(gpc / N);

  const float cx = coords[gpc * 3 + 0];
  const float cy = coords[gpc * 3 + 1];
  const float cz = coords[gpc * 3 + 2];

  const float us[3] = {cx, cy, cx};   // plane0:(x,y) plane1:(y,z) plane2:(x,z)
  const float vs[3] = {cy, cz, cz};

  int   o00[3], o01[3], o10[3], o11[3];
  float q00[3], q01[3], q10[3], q11[3];
  long  pbase[3];
#pragma unroll
  for (int pl = 0; pl < 3; ++pl) {
    float fx = (us[pl] + 1.0f) * 0.5f * (float)(RPLANE - 1);
    float fy = (vs[pl] + 1.0f) * 0.5f * (float)(RPLANE - 1);
    float x0f = floorf(fx), y0f = floorf(fy);
    float wx = fx - x0f,    wy = fy - y0f;
    int x0 = (int)x0f, y0 = (int)y0f;
    int x1 = min(x0 + 1, RPLANE - 1), y1 = min(y0 + 1, RPLANE - 1);
    x0 = max(min(x0, RPLANE - 1), 0);
    y0 = max(min(y0, RPLANE - 1), 0);
    o00[pl] = y0 * RPLANE + x0;  o01[pl] = y0 * RPLANE + x1;
    o10[pl] = y1 * RPLANE + x0;  o11[pl] = y1 * RPLANE + x1;
    q00[pl] = (1.0f - wx) * (1.0f - wy);  q01[pl] = wx * (1.0f - wy);
    q10[pl] = (1.0f - wx) * wy;           q11[pl] = wx * wy;
    pbase[pl] = (long)(bb * 3 + pl) * CDIM * (RPLANE * RPLANE);
  }
#pragma unroll 4
  for (int c = 0; c < CDIM; ++c) {
    float f = 1.0f;
#pragma unroll
    for (int pl = 0; pl < 3; ++pl) {
      const float* __restrict__ P = tp + pbase[pl] + (long)c * (RPLANE * RPLANE);
      float s = P[o00[pl]] * q00[pl] + P[o01[pl]] * q01[pl]
              + P[o10[pl]] * q10[pl] + P[o11[pl]] * q11[pl];
      f *= s;
    }
    sF[t * FPAD + c] = (_Float16)f;
  }
  __syncthreads();

  // ---- layer 0: [128,32] @ w0^T -> sH0[128,128]
  // wave owns row-tiles {2w, 2w+1}; one A-frag feeds 8 col-tiles (4 per group,
  // giving 4 independent back-to-back WMMAs to cover the f16 hazard window).
#pragma unroll
  for (int rr = 0; rr < 2; ++rr) {
    const int rt = wave * 2 + rr;
    const v16h a = load_a_frag(sF, rt * 16, FPAD, 0, m, lo);
#pragma unroll
    for (int cg = 0; cg < 2; ++cg) {
      v8f acc[4];
#pragma unroll
      for (int c = 0; c < 4; ++c) {
        const int ct = cg * 4 + c;
        v16h bm = load_b_frag(sW0, ct * 16, FPAD, 0, m, lo);
        v8f z = {};
        acc[c] = __builtin_amdgcn_wmma_f32_16x16x32_f16(false, a, false, bm,
                                                        (short)0, z, false, false);
      }
#pragma unroll
      for (int c = 0; c < 4; ++c) {
        const int ocol = (cg * 4 + c) * 16 + m;
        const float bias = sB0[ocol];
#pragma unroll
        for (int r = 0; r < 8; ++r) {
          int orow = rt * 16 + r + (lo ? 0 : 8);   // D: VGPR r -> M=r / M=8+r
          float v = fmaxf(acc[c][r] + bias, 0.0f);
          sH0[orow * HPAD + ocol] = (_Float16)v;
        }
      }
    }
  }
  __syncthreads();

  // ---- layer 1 (+ fused layer 2): wave owns all 8 col-tiles of its row-tiles,
  // so the 128-wide output dot-product reduces in-register across col-tiles,
  // then a 4-step shfl_xor across the 16-lane half, then one plain LDS store.
#pragma unroll
  for (int rr = 0; rr < 2; ++rr) {
    const int rt = wave * 2 + rr;
    float rowAcc[8] = {0.f, 0.f, 0.f, 0.f, 0.f, 0.f, 0.f, 0.f};
#pragma unroll
    for (int cg = 0; cg < 2; ++cg) {
      v8f acc[4] = {{}, {}, {}, {}};
#pragma unroll
      for (int ks = 0; ks < 4; ++ks) {
        const v16h a = load_a_frag(sH0, rt * 16, HPAD, ks * 32, m, lo);
#pragma unroll
        for (int c = 0; c < 4; ++c) {
          const int ct = cg * 4 + c;
          v16h bm = load_b_frag(sW1, ct * 16, HPAD, ks * 32, m, lo);
          acc[c] = __builtin_amdgcn_wmma_f32_16x16x32_f16(false, a, false, bm,
                                                          (short)0, acc[c],
                                                          false, false);
        }
      }
#pragma unroll
      for (int c = 0; c < 4; ++c) {
        const int ocol = (cg * 4 + c) * 16 + m;
        const float bias = sB1[ocol];
        const float wsc  = sW2[ocol];
#pragma unroll
        for (int r = 0; r < 8; ++r) {
          float v = fmaxf(acc[c][r] + bias, 0.0f);   // relu(h1)
          rowAcc[r] += v * wsc;                      // * w2[ocol]
        }
      }
    }
    // reduce across the 16 columns held by this 16-lane half
#pragma unroll
    for (int r = 0; r < 8; ++r) {
      float s = rowAcc[r];
      s += __shfl_xor(s, 1, 16);
      s += __shfl_xor(s, 2, 16);
      s += __shfl_xor(s, 4, 16);
      s += __shfl_xor(s, 8, 16);
      if (m == 0) sOut[rt * 16 + r + (lo ? 0 : 8)] = s;
    }
  }
  __syncthreads();

  if (live) out[gp] = sOut[t] + b2[0];
}

extern "C" void kernel_launch(void* const* d_in, const int* in_sizes, int n_in,
                              void* d_out, int out_size, void* d_ws, size_t ws_size,
                              hipStream_t stream) {
  const float* tp     = (const float*)d_in[0];
  const float* coords = (const float*)d_in[1];
  const float* w0 = (const float*)d_in[2];
  const float* b0 = (const float*)d_in[3];
  const float* w1 = (const float*)d_in[4];
  const float* b1 = (const float*)d_in[5];
  const float* w2 = (const float*)d_in[6];
  const float* b2 = (const float*)d_in[7];
  float* out = (float*)d_out;

  const long BN  = (long)in_sizes[1] / 3;                       // B*N points
  const int  Btp = in_sizes[0] / (3 * CDIM * RPLANE * RPLANE);  // batch count
  const int  N   = (int)(BN / (Btp > 0 ? Btp : 1));
  const int  nblocks = (int)((BN + TILE - 1) / TILE);

  triplane_mlp_kernel<<<nblocks, TILE, 0, stream>>>(
      tp, coords, w0, b0, w1, b1, w2, b2, out, N, BN);
}
